// AttentionLayer_28656021799393
// MI455X (gfx1250) — matrix operand order repaired and checked by reading, not compiled, not run on hardware
//
#include <hip/hip_runtime.h>

#define NB 4
#define TT 4096
#define SS 4096
#define DDIM 256

typedef __attribute__((ext_vector_type(16))) __bf16 v16bf;
typedef __attribute__((ext_vector_type(8)))  float  v8f;

union BF16x16 { v16bf v; uint4 u[2]; };

__device__ __forceinline__ float fast_exp2(float x) {
#if __has_builtin(__builtin_amdgcn_exp2f)
  return __builtin_amdgcn_exp2f(x);
#else
  return exp2f(x);
#endif
}

__device__ __forceinline__ float redmax16(float v) {
  v = fmaxf(v, __shfl_xor(v, 1, 32));
  v = fmaxf(v, __shfl_xor(v, 2, 32));
  v = fmaxf(v, __shfl_xor(v, 4, 32));
  v = fmaxf(v, __shfl_xor(v, 8, 32));
  return v;
}
__device__ __forceinline__ float redsum16(float v) {
  v += __shfl_xor(v, 1, 32);
  v += __shfl_xor(v, 2, 32);
  v += __shfl_xor(v, 4, 32);
  v += __shfl_xor(v, 8, 32);
  return v;
}

// --- CDNA5 async global->LDS copy (16B per lane), tracked by ASYNCcnt -------
__device__ __forceinline__ void async_copy16(void* lds, const void* g) {
  const unsigned ldsAddr = (unsigned)(size_t)lds;  // low 32 bits = LDS offset
  asm volatile("global_load_async_to_lds_b128 %0, %1, off"
               :: "v"(ldsAddr), "v"(g) : "memory");
}
__device__ __forceinline__ void wait_async0() {
#if __has_builtin(__builtin_amdgcn_s_wait_asynccnt)
  __builtin_amdgcn_s_wait_asynccnt(0);
#else
  asm volatile("s_wait_asynccnt 0x0" ::: "memory");
#endif
}

// ---------------------------------------------------------------------------
// Kernel 1: convert fp32 weights -> bf16 (one shot, 3 x 256 x 256)
// ---------------------------------------------------------------------------
__global__ void cvt_weights(const float* __restrict__ Wq,
                            const float* __restrict__ Wk,
                            const float* __restrict__ Wv,
                            __bf16* __restrict__ Wbf) {
  const int i = blockIdx.x * 256 + threadIdx.x;   // 0..65535
  Wbf[i]          = (__bf16)Wq[i];
  Wbf[65536 + i]  = (__bf16)Wk[i];
  Wbf[131072 + i] = (__bf16)Wv[i];
}

// ---------------------------------------------------------------------------
// Kernel 2: projection GEMM  Out[r, c] = sum_d A[r, d] * W[c, d] + bias[c]
// W slab (64 cols x 256) staged in LDS once; 8 waves reuse it.
// ---------------------------------------------------------------------------
__global__ void __launch_bounds__(256)
proj_gemm(const float* __restrict__ A, const __bf16* __restrict__ Wb,
          const float* __restrict__ bias, __bf16* __restrict__ Out) {
  const int tid  = threadIdx.x;
  const int wave = tid >> 5;
  const int lane = tid & 31;
  const int hn   = lane & 15;
  const int hi   = lane >> 4;
  const int kb   = hi * 8;
  const int koff = hi * 16;
  const int row0 = blockIdx.x * 128 + wave * 16;
  const int col0 = blockIdx.y * 64;

  __shared__ __bf16 Ws[64 * DDIM];   // 32 KB
  {
    const uint4* src = (const uint4*)(Wb + (size_t)col0 * DDIM);
#pragma unroll
    for (int it = 0; it < 8; ++it)
      async_copy16((uint4*)Ws + tid + it * 256, src + tid + it * 256);
    wait_async0();
  }
  __syncthreads();

  v8f acc[4] = {{}, {}, {}, {}};
  const float* arow = A + (size_t)(row0 + hn) * DDIM;

#pragma unroll
  for (int kc = 0; kc < 8; ++kc) {
    const float* ap = arow + kc * 32 + kb;
    float4 f0 = *(const float4*)(ap);
    float4 f1 = *(const float4*)(ap + 4);
    float4 f2 = *(const float4*)(ap + 16);
    float4 f3 = *(const float4*)(ap + 20);
    BF16x16 a;
    v16bf av = {(__bf16)f0.x, (__bf16)f0.y, (__bf16)f0.z, (__bf16)f0.w,
                (__bf16)f1.x, (__bf16)f1.y, (__bf16)f1.z, (__bf16)f1.w,
                (__bf16)f2.x, (__bf16)f2.y, (__bf16)f2.z, (__bf16)f2.w,
                (__bf16)f3.x, (__bf16)f3.y, (__bf16)f3.z, (__bf16)f3.w};
    a.v = av;
#pragma unroll
    for (int jj = 0; jj < 4; ++jj) {
      const __bf16* bp = Ws + (size_t)(jj * 16 + hn) * DDIM + kc * 32 + kb;
      BF16x16 b;
      b.u[0] = *(const uint4*)(bp);
      b.u[1] = *(const uint4*)(bp + 16);
      acc[jj] = __builtin_amdgcn_wmma_f32_16x16x32_bf16(
          false, a.v, false, b.v, (short)0, acc[jj], false, false);
    }
  }

#pragma unroll
  for (int jj = 0; jj < 4; ++jj) {
    const int   col  = col0 + jj * 16 + hn;
    const float bcol = bias[col];
#pragma unroll
    for (int r = 0; r < 8; ++r) {
      const int row = row0 + r + hi * 8;
      Out[(size_t)row * DDIM + col] = (__bf16)(acc[jj][r] + bcol);
    }
  }
}

// ---------------------------------------------------------------------------
// Kernel 3: causal flash attention, Bc = 64, double-buffered K/V staging.
// LDS: Q 64KB | K[2] 2x32KB | Vt[2] 2x32KB | P 16KB  = 208 KB (WGP has 320KB)
// ---------------------------------------------------------------------------
__global__ void __launch_bounds__(256)
flash_attn(const __bf16* __restrict__ Qb, const __bf16* __restrict__ Kb,
           const __bf16* __restrict__ Vb, float* __restrict__ Out) {
  const int n    = blockIdx.y;
  const int t0   = blockIdx.x * 128;
  const int tid  = threadIdx.x;
  const int wave = tid >> 5;
  const int lane = tid & 31;
  const int hn   = lane & 15;
  const int hi   = lane >> 4;
  const int kb   = hi * 8;
  const int koff = hi * 16;

  extern __shared__ char smem[];
  __bf16* Qs = (__bf16*)smem;                              // [128][256]
  __bf16* Pw = (__bf16*)(smem + 196608) + wave * 1024;     // [16][64] per wave
  // double buffers computed arithmetically (no LDS pointer arrays!)
  auto ksbuf = [&](int b) -> __bf16* { return (__bf16*)(smem + 65536  + (b << 15)); };
  auto vtbuf = [&](int b) -> __bf16* { return (__bf16*)(smem + 131072 + (b << 15)); };

  // ---- stage Q tile (async) ----
  {
    const uint4* src = (const uint4*)(Qb + ((size_t)n * TT + t0) * DDIM);
    uint4* dst = (uint4*)Qs;
#pragma unroll
    for (int it = 0; it < 16; ++it)
      async_copy16(dst + tid + it * 256, src + tid + it * 256);
  }

  // ---- staging helpers ----
  auto stage_k = [&](int buf, int s0) {
    const uint4* src = (const uint4*)(Kb + ((size_t)n * SS + s0) * DDIM);
    uint4* dst = (uint4*)ksbuf(buf);
#pragma unroll
    for (int it = 0; it < 8; ++it)
      async_copy16(dst + tid + it * 256, src + tid + it * 256);
  };
  auto stage_v = [&](int buf, int s0) {  // transpose while staging
    const unsigned short* vsrc =
        (const unsigned short*)(Vb + ((size_t)n * SS + s0) * DDIM);
    unsigned short* vt = (unsigned short*)vtbuf(buf);
#pragma unroll
    for (int it = 0; it < 16; ++it) {
      const int c  = tid + it * 256;   // 0..4095, 4 elems each
      const int s  = c >> 6;           // 0..63
      const int d0 = (c & 63) << 2;    // 0..252
      ushort4 d = *(const ushort4*)(vsrc + (size_t)s * DDIM + d0);
      vt[(d0 + 0) * 64 + s] = d.x;
      vt[(d0 + 1) * 64 + s] = d.y;
      vt[(d0 + 2) * 64 + s] = d.z;
      vt[(d0 + 3) * 64 + s] = d.w;
    }
  };

  stage_k(0, 0);
  stage_v(0, 0);
  wait_async0();
  __syncthreads();

  v8f O[16];
#pragma unroll
  for (int j = 0; j < 16; ++j) O[j] = (v8f){};
  float mrow[8], lrow[8];
#pragma unroll
  for (int r = 0; r < 8; ++r) { mrow[r] = -INFINITY; lrow[r] = 0.f; }

  const int   qrow  = wave * 16 + hn;
  const int   tminw = t0 + wave * 16;
  const int   tmaxw = tminw + 15;
  const float scl   = 0.0625f * 1.44269504088896f;  // 1/sqrt(256)*log2(e)
  const int   nS    = t0 + 128;

  for (int s0 = 0; s0 < nS; s0 += 64) {
    const int cur = (s0 >> 6) & 1;
    const int nxt = cur ^ 1;
    // prefetch next tile into the other buffer while computing this one
    if (s0 + 64 < nS) { stage_k(nxt, s0 + 64); stage_v(nxt, s0 + 64); }

    if (s0 <= tmaxw) {
      const __bf16* Ks = ksbuf(cur);
      const __bf16* Vt = vtbuf(cur);

      // ---- scores: kc outer so each Q fragment is loaded once, used 4x ----
      v8f sc4[4] = {{}, {}, {}, {}};
#pragma unroll
      for (int kc = 0; kc < 8; ++kc) {
        BF16x16 a;
        const __bf16* qp = Qs + qrow * DDIM + kc * 32;
        a.u[0] = *(const uint4*)(qp + kb);
        a.u[1] = *(const uint4*)(qp + 16 + kb);
#pragma unroll
        for (int nt = 0; nt < 4; ++nt) {
          const __bf16* kp = Ks + (nt * 16 + hn) * DDIM + kc * 32 + kb;
          BF16x16 b;
          b.u[0] = *(const uint4*)(kp);
          b.u[1] = *(const uint4*)(kp + 16);
          sc4[nt] = __builtin_amdgcn_wmma_f32_16x16x32_bf16(
              false, a.v, false, b.v, (short)0, sc4[nt], false, false);
        }
      }

      // ---- scale + causal mask ----
      float p[4][8];
      if (s0 + 63 <= tminw) {         // wave-uniform: fully unmasked tile
#pragma unroll
        for (int nt = 0; nt < 4; ++nt)
#pragma unroll
          for (int r = 0; r < 8; ++r) p[nt][r] = sc4[nt][r] * scl;
      } else {                        // diagonal tile: apply causal mask
#pragma unroll
        for (int nt = 0; nt < 4; ++nt) {
          const int scol = s0 + nt * 16 + hn;
#pragma unroll
          for (int r = 0; r < 8; ++r) {
            const int trow = tminw + r + hi * 8;
            p[nt][r] = (scol <= trow) ? sc4[nt][r] * scl : -INFINITY;
          }
        }
      }

      // ---- online softmax (base-2) ----
      float alpha[8];
#pragma unroll
      for (int r = 0; r < 8; ++r) {
        float mx = fmaxf(fmaxf(p[0][r], p[1][r]), fmaxf(p[2][r], p[3][r]));
        mx = redmax16(mx);
        const float mnew = fmaxf(mrow[r], mx);
        alpha[r] = fast_exp2(mrow[r] - mnew);
        mrow[r]  = mnew;
        p[0][r]  = fast_exp2(p[0][r] - mnew);
        p[1][r]  = fast_exp2(p[1][r] - mnew);
        p[2][r]  = fast_exp2(p[2][r] - mnew);
        p[3][r]  = fast_exp2(p[3][r] - mnew);
        lrow[r]  = lrow[r] * alpha[r] +
                   redsum16((p[0][r] + p[1][r]) + (p[2][r] + p[3][r]));
      }

      // rescale running O
#pragma unroll
      for (int j = 0; j < 16; ++j) {
#pragma unroll
        for (int r = 0; r < 8; ++r) O[j][r] *= alpha[r];
      }

      // ---- transpose P (C-layout -> A-layout) via wave-private LDS ----
#pragma unroll
      for (int nt = 0; nt < 4; ++nt) {
#pragma unroll
        for (int r = 0; r < 8; ++r)
          Pw[(r + hi * 8) * 64 + nt * 16 + hn] = (__bf16)p[nt][r];
      }
      BF16x16 pa0, pa1;
      {
        const __bf16* pp = Pw + hn * 64;
        pa0.u[0] = *(const uint4*)(pp + kb);
        pa0.u[1] = *(const uint4*)(pp + 16 + kb);
        pa1.u[0] = *(const uint4*)(pp + 32 + kb);
        pa1.u[1] = *(const uint4*)(pp + 48 + kb);
      }

      // ---- O += P(16x64) * V_tile(64x256) ----
#pragma unroll
      for (int j = 0; j < 16; ++j) {
        const __bf16* vp = Vt + (j * 16 + hn) * 64;
        BF16x16 b0, b1;
        b0.u[0] = *(const uint4*)(vp + kb);
        b0.u[1] = *(const uint4*)(vp + 16 + kb);
        b1.u[0] = *(const uint4*)(vp + 32 + kb);
        b1.u[1] = *(const uint4*)(vp + 48 + kb);
        O[j] = __builtin_amdgcn_wmma_f32_16x16x32_bf16(
            false, pa0.v, false, b0.v, (short)0, O[j], false, false);
        O[j] = __builtin_amdgcn_wmma_f32_16x16x32_bf16(
            false, pa1.v, false, b1.v, (short)0, O[j], false, false);
      }
    }

    wait_async0();   // next tile resident in LDS
    __syncthreads(); // all waves done with cur + staging of nxt visible
  }

  // ---- epilogue: divide by row sums, store fp32 ----
#pragma unroll
  for (int r = 0; r < 8; ++r) {
    const float inv  = 1.0f / lrow[r];
    const int   trow = t0 + wave * 16 + r + hi * 8;
    float* orow = Out + ((size_t)n * TT + trow) * DDIM;
#pragma unroll
    for (int j = 0; j < 16; ++j) orow[j * 16 + hn] = O[j][r] * inv;
  }
}

// ---------------------------------------------------------------------------
extern "C" void kernel_launch(void* const* d_in, const int* in_sizes, int n_in,
                              void* d_out, int out_size, void* d_ws, size_t ws_size,
                              hipStream_t stream) {
  const float* query = (const float*)d_in[0];
  const float* key_i = (const float*)d_in[1];
  const float* value = (const float*)d_in[2];
  // d_in[3] = attn_mask (causal tril) -- implemented analytically
  const float* Wq = (const float*)d_in[4];
  const float* bq = (const float*)d_in[5];
  const float* Wk = (const float*)d_in[6];
  const float* bk = (const float*)d_in[7];
  const float* Wv = (const float*)d_in[8];
  const float* bv = (const float*)d_in[9];

  char* ws = (char*)d_ws;
  __bf16* qbf = (__bf16*)(ws);             // 8 MB
  __bf16* kbf = (__bf16*)(ws + 8388608);   // 8 MB
  __bf16* vbf = (__bf16*)(ws + 16777216);  // 8 MB
  __bf16* Wbf = (__bf16*)(ws + 25165824);  // 384 KB

  cvt_weights<<<256, 256, 0, stream>>>(Wq, Wk, Wv, Wbf);

  proj_gemm<<<dim3(128, 4), 256, 0, stream>>>(query, Wbf,          bq, qbf);
  proj_gemm<<<dim3(128, 4), 256, 0, stream>>>(key_i, Wbf + 65536,  bk, kbf);
  proj_gemm<<<dim3(128, 4), 256, 0, stream>>>(value, Wbf + 131072, bv, vbf);

  // 32 query tiles of 128 rows per batch; 208 KB dynamic LDS
  flash_attn<<<dim3(32, 4), 256, 212992, stream>>>(qbf, kbf, vbf, (float*)d_out);
}
